// LSTMForecaster_62835371540790
// MI455X (gfx1250) — matrix operand order repaired and checked by reading, not compiled, not run on hardware
//
#include <hip/hip_runtime.h>
#include <hip/hip_bf16.h>

typedef _Float16 f16;
typedef __attribute__((ext_vector_type(16))) _Float16 v16h;
typedef __attribute__((ext_vector_type(8)))  _Float16 v8h;
typedef __attribute__((ext_vector_type(8)))  float    v8f;

// explicit global-address-space views (keeps loads as global_load_* after the
// LICM asm barrier, instead of decaying to flat_load_* which ties up DScnt)
typedef const __attribute__((address_space(1))) f16  gcf16;
typedef const __attribute__((address_space(1))) v16h gcv16h;

#define HDIM   256
#define FOURH  1024
#define SEQ    120
#define WARM   96
#define FCAST  24
#define TILE_B 16
#define HP     264      // padded h row stride (halves): 528B row -> conflict-free ds_load_b128
#define GP     1032     // padded gate row stride (halves)

// Convert W_hh to f16 (WMMA B operand source) and fuse the two bias vectors.
__global__ __launch_bounds__(256) void prep_kernel(const float* __restrict__ W_hh,
                                                   const float* __restrict__ b_ih,
                                                   const float* __restrict__ b_hh,
                                                   f16* __restrict__ w16,
                                                   float* __restrict__ btot) {
    int i = blockIdx.x * 256 + threadIdx.x;          // 1024*256 elements
    w16[i] = (f16)W_hh[i];
    if (i < FOURH) btot[i] = b_ih[i] + b_hh[i];
}

__global__ __launch_bounds__(256) void lstm_kernel(
    const float* __restrict__ x,    const float* __restrict__ h0,
    const float* __restrict__ c0,   const float* __restrict__ W_ih,
    const f16*   __restrict__ w16,  const float* __restrict__ btot,
    const float* __restrict__ fc_w, const float* __restrict__ fc_b,
    float* __restrict__ out)
{
    __shared__ f16    sh[TILE_B * HP];       // current h, f16
    __shared__ f16    sgate[TILE_B * GP];    // activated gates, f16
    __shared__ float2 sbw[FOURH];            // {bias, W_ih} per gate column
    __shared__ float  sfcw[HDIM];
    __shared__ float  sx[TILE_B];            // per-row scalar input for this step

    const int tid  = threadIdx.x;
    const int lane = tid & 31;
    const int wave = tid >> 5;               // 0..7, owns gate cols [wave*128, wave*128+128)
    const int bb   = blockIdx.x * TILE_B;    // first batch row of this tile

    // phase-2 mapping: thread owns (row r2, hidden cols j2..j2+15)
    const int r2 = tid >> 4;
    const int j2 = (tid & 15) * 16;

    // ---- one-time LDS / register init ----
    for (int k = tid; k < FOURH; k += 256) sbw[k] = make_float2(btot[k], W_ih[k]);
    for (int k = tid; k < HDIM;  k += 256) sfcw[k] = fc_w[k];
    float creg[16];
    #pragma unroll
    for (int jj = 0; jj < 16; ++jj) {
        size_t gi = (size_t)(bb + r2) * HDIM + j2 + jj;
        creg[jj] = c0[gi];
        sh[r2 * HP + j2 + jj] = (f16)h0[gi];
    }
    if (tid < TILE_B) sx[tid] = x[(size_t)(bb + tid) * SEQ + 0];
    __syncthreads();

    const int   m   = lane & 15;             // A row / B column within tile
    const int   hiK = lane >> 4;             // lane half selects K sub-range
    const float fcb = fc_b[0];

    const f16* abase = &sh[m * HP + hiK * 16];  // A frags in LDS (rewritten each step)

    for (int t = 0; t < SEQ; ++t) {
        // LICM barrier: forbid hoisting the (loop-invariant) weight loads out of
        // the t-loop (otherwise the compiler preloads ~2KB/lane of weights,
        // spills to scratch, and reloads from scratch instead of from L2).
        unsigned long long wq =
            (unsigned long long)(w16 + (size_t)(wave * 128 + m) * HDIM + hiK * 16);
        asm volatile("" : "+v"(wq));
        gcf16* wbase = (gcf16*)wq;           // global addr-space view

        // ---- build the 8 A fragments once per step (16x ds_load_b128) ----
        v16h afr[8];
        #pragma unroll
        for (int kt = 0; kt < 8; ++kt) {
            v8h lo = *(const v8h*)(abase + kt * 32);        // K = k0..k0+7
            v8h hi = *(const v8h*)(abase + kt * 32 + 8);    // K = k0+8..+15
            afr[kt] = __builtin_shufflevector(lo, hi,
                        0,1,2,3,4,5,6,7,8,9,10,11,12,13,14,15);
        }
        float xr[8];
        #pragma unroll
        for (int r = 0; r < 8; ++r) xr[r] = sx[hiK * 8 + r];

        // ---- this wave's 8 N-tiles of 16 gate columns ----
        #pragma unroll
        for (int nt = 0; nt < 8; ++nt) {
            const int col = wave * 128 + nt * 16 + m;      // global gate column
            const float2 bw = sbw[col];                    // one ds_load_b64
            v8f acc;
            #pragma unroll
            for (int r = 0; r < 8; ++r) acc[r] = fmaf(xr[r], bw.y, bw.x); // bias + x*W_ih

            gcf16* wp = wbase + (size_t)nt * 16 * HDIM;
            #pragma unroll
            for (int kt = 0; kt < 8; ++kt) {
                // B fragment: lane = column, 16 contiguous K halves (32B aligned)
                v16h b = *(gcv16h*)(wp + kt * 32);
                acc = __builtin_amdgcn_wmma_f32_16x16x32_f16(
                          false, afr[kt], false, b, (short)0, acc, false, false);
            }

            // branchless activation: a*rcp(1+exp(-k*v)) + c
            //   sigmoid: k=1,a=1,c=0   tanh: k=2,a=2,c=-1  (tanh(x)=2*sig(2x)-1)
            const int   gate = (wave * 128 + nt * 16) >> 8;  // 0=i 1=f 2=g 3=o
            const float kk = (gate == 2) ? 2.0f : 1.0f;
            const float cc = (gate == 2) ? -1.0f : 0.0f;
            #pragma unroll
            for (int r = 0; r < 8; ++r) {
                float s = __builtin_amdgcn_rcpf(1.0f + __expf(-kk * acc[r]));
                float v = fmaf(kk, s, cc);
                int row = r + hiK * 8;                     // ISA C-layout row
                sgate[row * GP + col] = (f16)v;
            }
        }
        __syncthreads();

        // ---- phase 2: cell/hidden update + FC head + next input ----
        float pd = 0.0f;
        #pragma unroll
        for (int jj = 0; jj < 16; ++jj) {
            int j = j2 + jj;
            float ig = (float)sgate[r2 * GP + j];
            float fg = (float)sgate[r2 * GP + 256 + j];
            float gg = (float)sgate[r2 * GP + 512 + j];
            float og = (float)sgate[r2 * GP + 768 + j];
            float c  = fg * creg[jj] + ig * gg;
            creg[jj] = c;
            float th = fmaf(2.0f, __builtin_amdgcn_rcpf(1.0f + __expf(-2.0f * c)), -1.0f);
            float hv = og * th;
            sh[r2 * HP + j] = (f16)hv;
            pd += hv * sfcw[j];
        }

        if (t + 1 < WARM) {
            // warm phase: next input comes from x
            if (tid < TILE_B) sx[tid] = x[(size_t)(bb + tid) * SEQ + (t + 1)];
        } else {
            // forecast head: the 16 partials of row r2 live in one wave ->
            // reduce with lane shuffles, no LDS / extra barrier needed
            pd += __shfl_xor(pd, 1, 32);
            pd += __shfl_xor(pd, 2, 32);
            pd += __shfl_xor(pd, 4, 32);
            pd += __shfl_xor(pd, 8, 32);
            if ((tid & 15) == 0) {
                float y = pd + fcb;
                if (t >= WARM) out[(size_t)(bb + r2) * FCAST + (t - WARM)] = y;
                sx[r2] = y;                                // feed y back as input
            }
        }
        __syncthreads();
    }
}

extern "C" void kernel_launch(void* const* d_in, const int* in_sizes, int n_in,
                              void* d_out, int out_size, void* d_ws, size_t ws_size,
                              hipStream_t stream) {
    const float* x    = (const float*)d_in[0];
    const float* h0   = (const float*)d_in[1];
    const float* c0   = (const float*)d_in[2];
    const float* W_ih = (const float*)d_in[3];
    const float* W_hh = (const float*)d_in[4];
    const float* b_ih = (const float*)d_in[5];
    const float* b_hh = (const float*)d_in[6];
    const float* fc_w = (const float*)d_in[7];
    const float* fc_b = (const float*)d_in[8];
    float* out = (float*)d_out;

    const int B = in_sizes[1] / HDIM;        // 2048

    f16*   w16  = (f16*)d_ws;
    float* btot = (float*)((char*)d_ws + (size_t)FOURH * HDIM * sizeof(f16));

    prep_kernel<<<(FOURH * HDIM) / 256, 256, 0, stream>>>(W_hh, b_ih, b_hh, w16, btot);
    lstm_kernel<<<B / TILE_B, 256, 0, stream>>>(x, h0, c0, W_ih, w16, btot,
                                                fc_w, fc_b, out);
}
